// BasicCrossAttention_83288005804684
// MI455X (gfx1250) — matrix operand order repaired and checked by reading, not compiled, not run on hardware
//
#include <hip/hip_runtime.h>

typedef __attribute__((ext_vector_type(16))) _Float16 v16h;
typedef __attribute__((ext_vector_type(8)))  _Float16 v8h;
typedef __attribute__((ext_vector_type(4)))  _Float16 v4h;
typedef __attribute__((ext_vector_type(8)))  float    v8f;

union Frag { v16h v; v8h h[2]; };

// ---------------------------------------------------------------- constants
#define BB     2
#define NN     2048
#define CC     1024
#define HH     16
#define DD     64
#define ROWS   (BB*NN)          // 4096
#define SCALE  0.125f
#define LN_EPS 1e-5f

// ---------------------------------------------------------------- helpers
__device__ __forceinline__ float redMax16(float v) {
    v = fmaxf(v, __shfl_xor(v, 1, 32));
    v = fmaxf(v, __shfl_xor(v, 2, 32));
    v = fmaxf(v, __shfl_xor(v, 4, 32));
    v = fmaxf(v, __shfl_xor(v, 8, 32));
    return v;
}
__device__ __forceinline__ float redSum16(float v) {
    v += __shfl_xor(v, 1, 32);
    v += __shfl_xor(v, 2, 32);
    v += __shfl_xor(v, 4, 32);
    v += __shfl_xor(v, 8, 32);
    return v;
}
__device__ __forceinline__ float redSum32(float v) {
    v += __shfl_xor(v, 1, 32);
    v += __shfl_xor(v, 2, 32);
    v += __shfl_xor(v, 4, 32);
    v += __shfl_xor(v, 8, 32);
    v += __shfl_xor(v, 16, 32);
    return v;
}
__device__ __forceinline__ v8f wmma16(const v16h& a, const v16h& b, const v8f& c) {
    return __builtin_amdgcn_wmma_f32_16x16x32_f16(false, a, false, b, (short)0, c, false, false);
}

// ---------------------------------------------------------------- f32 -> f16
__global__ __launch_bounds__(256) void cvt_f32_f16(const float* __restrict__ s,
                                                   _Float16* __restrict__ d, int n4) {
    int i = blockIdx.x * blockDim.x + threadIdx.x;
    if (i >= n4) return;
    float4 f = ((const float4*)s)[i];
    v4h h;
    h[0] = (_Float16)f.x; h[1] = (_Float16)f.y;
    h[2] = (_Float16)f.z; h[3] = (_Float16)f.w;
    ((v4h*)d)[i] = h;
}

// ---------------------------------------------------------------- GEMM  Y = X @ W^T
// X: [ROWS, 1024] f16 row-major.  W: [1024, 1024] f16 row-major (torch Linear weight).
// mode 0: f32 out [ROWS,1024]; mode 1: f16 out transposed to [B,H,D,M]; mode 2: f32+bias.
// Block 256 thr = 8 waves (4 M x 2 N), wave tile 32x64 -> block tile 128x128.
__global__ __launch_bounds__(256)
void gemm_xwT(const _Float16* __restrict__ X, const _Float16* __restrict__ W,
              float* __restrict__ Yf, _Float16* __restrict__ Yvt,
              const float* __restrict__ bias, float* __restrict__ Yout, int mode) {
    const int lane  = threadIdx.x & 31;
    const int wave  = threadIdx.x >> 5;
    const int hg    = lane >> 4;
    const int ln15  = lane & 15;
    const int waveM = wave >> 1;                 // 0..3
    const int waveN = wave & 1;                  // 0..1
    const int rowBase = blockIdx.x * 128 + waveM * 32;
    const int colBase = blockIdx.y * 128 + waveN * 64;

    v8f c[2][4] = {};

    for (int cb = 0; cb < CC; cb += 32) {
        Frag a[2], b[4];
#pragma unroll
        for (int mi = 0; mi < 2; ++mi) {
            const _Float16* p = X + (size_t)(rowBase + 16 * mi + ln15) * CC + cb;
            a[mi].h[0] = *(const v8h*)(p + 16 * hg);
            a[mi].h[1] = *(const v8h*)(p + 16 * hg + 8);
        }
#pragma unroll
        for (int ni = 0; ni < 4; ++ni) {
            const _Float16* p = W + (size_t)(colBase + 16 * ni + ln15) * CC + cb + 16 * hg;
            b[ni].h[0] = *(const v8h*)(p);
            b[ni].h[1] = *(const v8h*)(p + 8);
        }
#pragma unroll
        for (int mi = 0; mi < 2; ++mi)
#pragma unroll
            for (int ni = 0; ni < 4; ++ni)
                c[mi][ni] = wmma16(a[mi].v, b[ni].v, c[mi][ni]);
    }

#pragma unroll
    for (int mi = 0; mi < 2; ++mi)
#pragma unroll
        for (int ni = 0; ni < 4; ++ni)
#pragma unroll
            for (int r = 0; r < 8; ++r) {
                int row = rowBase + 16 * mi + r + 8 * hg;
                int col = colBase + 16 * ni + ln15;
                float val = c[mi][ni][r];
                if (mode == 0) {
                    Yf[(size_t)row * CC + col] = val;
                } else if (mode == 1) {
                    int bi = row >> 11, m = row & 2047;
                    int hh = col >> 6,  dd = col & 63;
                    Yvt[(size_t)(((bi * HH + hh) * DD + dd)) * NN + m] = (_Float16)val;
                } else {
                    Yout[(size_t)row * CC + col] = val + bias[col];
                }
            }
}

// ---------------------------------------------------------------- per-head LayerNorm
// q/k are [ROWS*16] head-rows of 64 contiguous floats. One wave per head-row.
__global__ __launch_bounds__(256)
void qk_layernorm(const float* __restrict__ qf, const float* __restrict__ kf,
                  _Float16* __restrict__ qh, _Float16* __restrict__ kh,
                  const float* __restrict__ g, const float* __restrict__ bia) {
    const int gw   = (blockIdx.x * blockDim.x + threadIdx.x) >> 5;
    const int lane = threadIdx.x & 31;
    const int NHR  = ROWS * HH;                 // 65536 head-rows per tensor
    const bool isK = gw >= NHR;
    const int row  = isK ? gw - NHR : gw;
    const float* src = isK ? kf : qf;
    _Float16*    dst = isK ? kh : qh;

    float2 x = *(const float2*)(src + (size_t)row * 64 + lane * 2);
    float mu = redSum32(x.x + x.y) * (1.0f / 64.0f);
    float d0 = x.x - mu, d1 = x.y - mu;
    float var = redSum32(d0 * d0 + d1 * d1) * (1.0f / 64.0f);
    float rs  = rsqrtf(var + LN_EPS);
    float sc  = isK ? 1.0f : SCALE;
    float y0  = (d0 * rs * g[2 * lane]     + bia[2 * lane])     * sc;
    float y1  = (d1 * rs * g[2 * lane + 1] + bia[2 * lane + 1]) * sc;
    dst[(size_t)row * 64 + 2 * lane]     = (_Float16)y0;
    dst[(size_t)row * 64 + 2 * lane + 1] = (_Float16)y1;
}

// ---------------------------------------------------------------- flash attention
// One wave handles one (b, h, 16-row q tile). 4 waves / block.
__global__ __launch_bounds__(128)
void flash_attn(const _Float16* __restrict__ qh, const _Float16* __restrict__ kh,
                const _Float16* __restrict__ vt, _Float16* __restrict__ oh) {
    __shared__ __attribute__((aligned(16))) _Float16 lds[4][16][40];

    const int lane = threadIdx.x & 31;
    const int w    = threadIdx.x >> 5;
    const int hg   = lane >> 4;
    const int ln15 = lane & 15;

    const int tile  = blockIdx.x * 4 + w;       // 0..4095
    const int b     = tile >> 11;               // 2048 tiles per batch
    const int h     = (tile >> 7) & 15;
    const int nbase = (tile & 127) * 16;

    // Q fragments (d 0..31, 32..63)
    Frag qa[2];
    {
        const _Float16* p = qh + (size_t)(b * NN + nbase + ln15) * CC + h * DD;
#pragma unroll
        for (int f = 0; f < 2; ++f) {
            qa[f].h[0] = *(const v8h*)(p + 32 * f + 16 * hg);
            qa[f].h[1] = *(const v8h*)(p + 32 * f + 16 * hg + 8);
        }
    }

    v8f o[4] = {};
    float mI[8], lI[8];
#pragma unroll
    for (int r = 0; r < 8; ++r) { mI[r] = -1e30f; lI[r] = 0.0f; }

    const _Float16* kbase = kh + (size_t)(b * NN) * CC + h * DD;
    const _Float16* vbase = vt + (size_t)((b * HH + h) * DD) * NN;

    for (int mt = 0; mt < NN; mt += 32) {
        // S = Q @ K^T   (two 16-col sub-tiles)
        v8f s0 = {}, s1 = {};
        {
            const _Float16* p = kbase + (size_t)(mt + ln15) * CC + 16 * hg;
            Frag b0, b1;
            b0.h[0] = *(const v8h*)(p);      b0.h[1] = *(const v8h*)(p + 8);
            b1.h[0] = *(const v8h*)(p + 32); b1.h[1] = *(const v8h*)(p + 40);
            s0 = wmma16(qa[0].v, b0.v, s0);
            s0 = wmma16(qa[1].v, b1.v, s0);
        }
        {
            const _Float16* p = kbase + (size_t)(mt + 16 + ln15) * CC + 16 * hg;
            Frag b0, b1;
            b0.h[0] = *(const v8h*)(p);      b0.h[1] = *(const v8h*)(p + 8);
            b1.h[0] = *(const v8h*)(p + 32); b1.h[1] = *(const v8h*)(p + 40);
            s1 = wmma16(qa[0].v, b0.v, s1);
            s1 = wmma16(qa[1].v, b1.v, s1);
        }

        // online softmax
        float p0[8], p1[8];
#pragma unroll
        for (int r = 0; r < 8; ++r) {
            float vmax = redMax16(fmaxf(s0[r], s1[r]));
            float mn   = fmaxf(mI[r], vmax);
            float al   = __expf(mI[r] - mn);
            p0[r] = __expf(s0[r] - mn);
            p1[r] = __expf(s1[r] - mn);
            float rsum = redSum16(p0[r] + p1[r]);
            lI[r] = lI[r] * al + rsum;
            mI[r] = mn;
#pragma unroll
            for (int dt = 0; dt < 4; ++dt) o[dt][r] *= al;
        }

        // re-layout P (D layout -> A layout) through per-wave LDS
#pragma unroll
        for (int r = 0; r < 8; ++r) {
            lds[w][r + 8 * hg][ln15]      = (_Float16)p0[r];
            lds[w][r + 8 * hg][16 + ln15] = (_Float16)p1[r];
        }
        Frag pa;
        pa.h[0] = *(const v8h*)(&lds[w][ln15][16 * hg]);
        pa.h[1] = *(const v8h*)(&lds[w][ln15][16 * hg + 8]);

        // O += P @ V
#pragma unroll
        for (int dt = 0; dt < 4; ++dt) {
            const _Float16* vp = vbase + (size_t)(16 * dt + ln15) * NN + mt + 16 * hg;
            Frag vb;
            vb.h[0] = *(const v8h*)(vp);
            vb.h[1] = *(const v8h*)(vp + 8);
            o[dt] = wmma16(pa.v, vb.v, o[dt]);
        }
    }

    // normalize + store f16 [B,N,C]
    float inv[8];
#pragma unroll
    for (int r = 0; r < 8; ++r) inv[r] = 1.0f / lI[r];
#pragma unroll
    for (int dt = 0; dt < 4; ++dt)
#pragma unroll
        for (int r = 0; r < 8; ++r) {
            int row = b * NN + nbase + r + 8 * hg;
            int col = h * DD + 16 * dt + ln15;
            oh[(size_t)row * CC + col] = (_Float16)(o[dt][r] * inv[r]);
        }
}

// ---------------------------------------------------------------- launch
extern "C" void kernel_launch(void* const* d_in, const int* in_sizes, int n_in,
                              void* d_out, int out_size, void* d_ws, size_t ws_size,
                              hipStream_t stream) {
    const float* x1  = (const float*)d_in[0];
    const float* x2  = (const float*)d_in[1];
    const float* Wq  = (const float*)d_in[2];
    const float* Wk  = (const float*)d_in[3];
    const float* Wv  = (const float*)d_in[4];
    const float* Wp  = (const float*)d_in[5];
    const float* bp  = (const float*)d_in[6];
    const float* lng = (const float*)d_in[7];
    const float* lnb = (const float*)d_in[8];

    char* ws = (char*)d_ws;
    const size_t MB = 1u << 20;
    float*    qf32 = (float*)(ws + 0 * MB);      // 16 MiB
    float*    kf32 = (float*)(ws + 16 * MB);     // 16 MiB
    _Float16* qh   = (_Float16*)(ws + 32 * MB);  // 8 MiB
    _Float16* kh   = (_Float16*)(ws + 40 * MB);  // 8 MiB
    _Float16* vt   = (_Float16*)(ws + 48 * MB);  // 8 MiB  [B,H,D,M]
    _Float16* oh   = (_Float16*)(ws + 56 * MB);  // 8 MiB
    _Float16* x1h  = (_Float16*)(ws + 64 * MB);  // 8 MiB
    _Float16* x2h  = (_Float16*)(ws + 72 * MB);  // 8 MiB
    _Float16* wqh  = (_Float16*)(ws + 80 * MB);  // 2 MiB each
    _Float16* wkh  = (_Float16*)(ws + 82 * MB);
    _Float16* wvh  = (_Float16*)(ws + 84 * MB);
    _Float16* wph  = (_Float16*)(ws + 86 * MB);

    const int nX  = ROWS * CC;      // 4194304
    const int nW  = CC * CC;        // 1048576
    cvt_f32_f16<<<nX / 1024, 256, 0, stream>>>(x1, x1h, nX / 4);
    cvt_f32_f16<<<nX / 1024, 256, 0, stream>>>(x2, x2h, nX / 4);
    cvt_f32_f16<<<nW / 1024, 256, 0, stream>>>(Wq, wqh, nW / 4);
    cvt_f32_f16<<<nW / 1024, 256, 0, stream>>>(Wk, wkh, nW / 4);
    cvt_f32_f16<<<nW / 1024, 256, 0, stream>>>(Wv, wvh, nW / 4);
    cvt_f32_f16<<<nW / 1024, 256, 0, stream>>>(Wp, wph, nW / 4);

    dim3 ggrid(ROWS / 128, CC / 128);            // 32 x 8
    gemm_xwT<<<ggrid, 256, 0, stream>>>(x1h, wqh, qf32, nullptr, nullptr, nullptr, 0);
    gemm_xwT<<<ggrid, 256, 0, stream>>>(x2h, wkh, kf32, nullptr, nullptr, nullptr, 0);
    gemm_xwT<<<ggrid, 256, 0, stream>>>(x2h, wvh, nullptr, vt, nullptr, nullptr, 1);

    // 2 * 65536 head-rows, one wave per row, 8 waves per 256-thread block
    qk_layernorm<<<(2 * ROWS * HH) / 8, 256, 0, stream>>>(qf32, kf32, qh, kh, lng, lnb);

    flash_attn<<<(BB * HH * (NN / 16)) / 4, 128, 0, stream>>>(qh, kh, vt, oh);

    gemm_xwT<<<ggrid, 256, 0, stream>>>(oh, wph, nullptr, nullptr, bp, (float*)d_out, 2);
}
